// Attention_46600395162185
// MI455X (gfx1250) — matrix operand order repaired and checked by reading, not compiled, not run on hardware
//
#include <hip/hip_runtime.h>
#include <hip/hip_bf16.h>
#include <stddef.h>

#define B_    8
#define LQ_   256
#define LK_   256
#define HID_  128
#define NH_   8
#define HD_   16
#define D_    128
#define SCALE_ 0.25f   // HD^-0.5
#define BPAD_ 40       // halves per 32-K column block (80B: 16B-aligned, bank-spread)

typedef __attribute__((ext_vector_type(16))) _Float16 v16h;
typedef __attribute__((ext_vector_type(8)))  float    v8f;

union HFrag { v16h v; uint4 q[2]; _Float16 h[16]; };

// ---- WMMA fragment loaders (wave32 layouts per CDNA5 ISA 7.12.2) -----------

// A: 16x32 f16 tile at (row0,k0), row-major LDS, stride lda halves.
// lanes 0-15: row=lh, K: j<8 -> k0+j, j>=8 -> k0+16+(j-8)
// lanes16-31: row=lh, K: j<8 -> k0+8+j, j>=8 -> k0+24+(j-8)
// Two contiguous 8-half runs -> 2x ds_load_b128.
__device__ __forceinline__ v16h load_a_frag(const _Float16* lds, int row0, int k0,
                                            int lda, int lane) {
  int half = lane >> 4, lh = lane & 15;
  const _Float16* p = lds + (row0 + lh) * lda + k0 + half * 8;
  HFrag f;
  f.q[0] = *(const uint4*)(p);
  f.q[1] = *(const uint4*)(p + 16);
  return f.v;
}

// B packed layout: Bp[((k>>5)*Ncols + n)*BPAD_ + (k&31)].
// Fragment element j = B[k0 + half*8 + j (j<8) / k0 + 16 + half*8 + (j-8) (j>=8), n0+lh].
__device__ __forceinline__ v16h load_b_packed(const _Float16* bp, int kb, int n0,
                                              int Ncols, int lane) {
  int half = lane >> 4, lh = lane & 15;
  const _Float16* p = bp + (size_t)(kb * Ncols + n0 + lh) * BPAD_ + half * 8;
  HFrag f;
  f.q[0] = *(const uint4*)(p);
  f.q[1] = *(const uint4*)(p + 16);
  return f.v;
}

__device__ __forceinline__ v8f wmma_gemm_tile(const _Float16* A, int lda,
                                              const _Float16* Bp, int Ncols,
                                              int row0, int n0, int K, int lane) {
  v8f c = {0.f, 0.f, 0.f, 0.f, 0.f, 0.f, 0.f, 0.f};
#pragma unroll
  for (int k0 = 0; k0 < K; k0 += 32) {
    v16h a = load_a_frag(A, row0, k0, lda, lane);
    v16h b = load_b_packed(Bp, k0 >> 5, n0, Ncols, lane);
    c = __builtin_amdgcn_wmma_f32_16x16x32_f16(false, a, false, b, (short)0, c,
                                               false, false);
  }
  return c;
}

// stage a KxN row-major fp32 weight into packed-B f16 LDS
__device__ __forceinline__ void stage_w(_Float16* wb, const float* W, int KN, int tid) {
  for (int i = tid; i < KN; i += 256) {
    int k = i >> 7, n = i & 127;   // N == 128 always here
    wb[(size_t)((k >> 5) * 128 + n) * BPAD_ + (k & 31)] = (_Float16)W[i];
  }
}

// ---- Kernel 1: LN(x_src) -> k = xs@Wk, v = xs@Wv + bv ----------------------
// smem: xs 64x136 h (17408) + wb packed 4*128*40 h (40960) = 58368
__global__ void k_proj_src(const float* __restrict__ xsrc,
                           const float* __restrict__ Wk, const float* __restrict__ Wv,
                           const float* __restrict__ bv,
                           const float* __restrict__ lng, const float* __restrict__ lnb,
                           float* __restrict__ k_ws, float* __restrict__ v_ws) {
  extern __shared__ char smem[];
  _Float16* xs = (_Float16*)smem;              // 64 x 136
  _Float16* wb = (_Float16*)(smem + 17408);    // packed B
  int row0 = blockIdx.x * 64;
  int tid = threadIdx.x, lane = tid & 31, wave = tid >> 5;

  if (tid < 64) {
    const float* xr = xsrc + (size_t)(row0 + tid) * HID_;
    float m = 0.f;
    for (int i = 0; i < HID_; ++i) m += xr[i];
    m *= (1.f / HID_);
    float v2 = 0.f;
    for (int i = 0; i < HID_; ++i) { float d = xr[i] - m; v2 += d * d; }
    float inv = rsqrtf(v2 * (1.f / HID_) + 1e-5f);
    _Float16* dst = xs + tid * 136;
    for (int i = 0; i < HID_; ++i)
      dst[i] = (_Float16)((xr[i] - m) * inv * lng[i] + lnb[i]);
  }
  stage_w(wb, Wk, 128 * 128, tid);
  __syncthreads();

  for (int t = 0; t < 4; ++t) {
    int id = wave * 4 + t, mt = id >> 3, nt = id & 7;
    v8f c = wmma_gemm_tile(xs, 136, wb, 128, mt * 16, nt * 16, 128, lane);
    int half = lane >> 4, lh = lane & 15, col = nt * 16 + lh;
#pragma unroll
    for (int i = 0; i < 8; ++i)
      k_ws[(size_t)(row0 + mt * 16 + half * 8 + i) * D_ + col] = c[i];
  }
  __syncthreads();
  stage_w(wb, Wv, 128 * 128, tid);
  __syncthreads();
  for (int t = 0; t < 4; ++t) {
    int id = wave * 4 + t, mt = id >> 3, nt = id & 7;
    v8f c = wmma_gemm_tile(xs, 136, wb, 128, mt * 16, nt * 16, 128, lane);
    int half = lane >> 4, lh = lane & 15, col = nt * 16 + lh;
#pragma unroll
    for (int i = 0; i < 8; ++i)
      v_ws[(size_t)(row0 + mt * 16 + half * 8 + i) * D_ + col] = c[i] + bv[col];
  }
}

// ---- Kernel 2: LN(x_dst) -> q = xd@Wq+bq, s = xd@Ws+bs, store xd -----------
__global__ void k_proj_dst(const float* __restrict__ xdst,
                           const float* __restrict__ Wq, const float* __restrict__ bq,
                           const float* __restrict__ Ws, const float* __restrict__ bs,
                           const float* __restrict__ lng, const float* __restrict__ lnb,
                           float* __restrict__ q_ws, float* __restrict__ s_ws,
                           float* __restrict__ xd_ws) {
  extern __shared__ char smem[];
  _Float16* xs = (_Float16*)smem;
  _Float16* wb = (_Float16*)(smem + 17408);
  int row0 = blockIdx.x * 64;
  int tid = threadIdx.x, lane = tid & 31, wave = tid >> 5;

  if (tid < 64) {
    const float* xr = xdst + (size_t)(row0 + tid) * HID_;
    float m = 0.f;
    for (int i = 0; i < HID_; ++i) m += xr[i];
    m *= (1.f / HID_);
    float v2 = 0.f;
    for (int i = 0; i < HID_; ++i) { float d = xr[i] - m; v2 += d * d; }
    float inv = rsqrtf(v2 * (1.f / HID_) + 1e-5f);
    _Float16* dst = xs + tid * 136;
    float* xo = xd_ws + (size_t)(row0 + tid) * HID_;
    for (int i = 0; i < HID_; ++i) {
      float xn = (xr[i] - m) * inv * lng[i] + lnb[i];
      dst[i] = (_Float16)xn;
      xo[i] = xn;
    }
  }
  stage_w(wb, Wq, 128 * 128, tid);
  __syncthreads();
  for (int t = 0; t < 4; ++t) {
    int id = wave * 4 + t, mt = id >> 3, nt = id & 7;
    v8f c = wmma_gemm_tile(xs, 136, wb, 128, mt * 16, nt * 16, 128, lane);
    int half = lane >> 4, lh = lane & 15, col = nt * 16 + lh;
#pragma unroll
    for (int i = 0; i < 8; ++i)
      q_ws[(size_t)(row0 + mt * 16 + half * 8 + i) * D_ + col] = c[i] + bq[col];
  }
  __syncthreads();
  stage_w(wb, Ws, 128 * 128, tid);
  __syncthreads();
  for (int t = 0; t < 4; ++t) {
    int id = wave * 4 + t, mt = id >> 3, nt = id & 7;
    v8f c = wmma_gemm_tile(xs, 136, wb, 128, mt * 16, nt * 16, 128, lane);
    int half = lane >> 4, lh = lane & 15, col = nt * 16 + lh;
#pragma unroll
    for (int i = 0; i < 8; ++i)
      s_ws[(size_t)(row0 + mt * 16 + half * 8 + i) * D_ + col] = c[i] + bs[col];
  }
}

// ---- Kernel 3: fused relational attention, one WG per (b, q) ---------------
// smem layout (170496 B):
//  rn   256x136 h @0       (69632)   row-major LN(r)
//  rnT  128x280 h @69632   (71680)   transposed copy for rn^T GEMM
//  wqB  packed 4*16*40 h @141312 (5120)
//  scB  packed 8*16*40 h @146432 (10240)
//  scF  256x8 f  @156672 (8192)
//  srF  128x8 f  @164864 (4096)
//  qv @168960, grs @169472, brs @169984  (512 each)
__global__ void k_attn(const float* __restrict__ rten,
                       const float* __restrict__ q_ws, const float* __restrict__ k_ws,
                       const float* __restrict__ v_ws,
                       const float* __restrict__ Wkr, const float* __restrict__ Wvr,
                       const float* __restrict__ bvr,
                       const float* __restrict__ lng, const float* __restrict__ lnb,
                       float* __restrict__ attn_ws) {
  extern __shared__ char smem[];
  _Float16* rn  = (_Float16*)smem;
  _Float16* rnT = (_Float16*)(smem + 69632);
  _Float16* wqB = (_Float16*)(smem + 141312);
  _Float16* scB = (_Float16*)(smem + 146432);
  float* scF = (float*)(smem + 156672);
  float* srF = (float*)(smem + 164864);
  float* qv  = (float*)(smem + 168960);
  float* grs = (float*)(smem + 169472);
  float* brs = (float*)(smem + 169984);

  int tid = threadIdx.x, lane = tid & 31, wave = tid >> 5;
  int bIdx = blockIdx.x >> 8;
  int iq   = blockIdx.x & 255;

  const float* qrow = q_ws + ((size_t)bIdx * LQ_ + iq) * D_;
  if (tid < 128) { qv[tid] = qrow[tid]; grs[tid] = lng[tid]; brs[tid] = lnb[tid]; }
  __syncthreads();

  // wq[c,h] = sum_d Wkr[c, h*16+d] * q[h*16+d] -> packed B (cols 8..15 zero)
  for (int i = tid; i < 1024; i += 256) {
    int c = i >> 3, h = i & 7;
    const float* wp = Wkr + c * D_ + h * HD_;
    const float* qp = qv + h * HD_;
    float acc = 0.f;
#pragma unroll
    for (int d = 0; d < HD_; ++d) acc += wp[d] * qp[d];
    int kb = c >> 5, jj = c & 31;
    wqB[(size_t)(kb * 16 + h) * BPAD_ + jj] = (_Float16)acc;
    wqB[(size_t)(kb * 16 + 8 + h) * BPAD_ + jj] = (_Float16)0.f;
  }
  // base scores: q . k -> scF
  const float* kbase = k_ws + (size_t)bIdx * LK_ * D_;
  for (int i = tid; i < 2048; i += 256) {
    int kk = i >> 3, h = i & 7;
    const float* kp = kbase + kk * D_ + h * HD_;
    const float* qp = qv + h * HD_;
    float acc = 0.f;
#pragma unroll
    for (int d = 0; d < HD_; ++d) acc += kp[d] * qp[d];
    scF[kk * 8 + h] = acc;
  }
  // LayerNorm of r rows -> rn (row-major) and rnT (transposed)
  {
    const float* rr = rten + (((size_t)bIdx * LQ_ + iq) * LK_ + tid) * HID_;
    __builtin_prefetch(v_ws + (size_t)bIdx * LK_ * D_ + (size_t)tid * D_, 0, 0);
    float m = 0.f;
    for (int i = 0; i < HID_; ++i) m += rr[i];
    m *= (1.f / HID_);
    float v2 = 0.f;
    for (int i = 0; i < HID_; ++i) { float d = rr[i] - m; v2 += d * d; }
    float inv = rsqrtf(v2 * (1.f / HID_) + 1e-5f);
    _Float16* dst = rn + tid * 136;
    for (int i = 0; i < HID_; ++i) {
      _Float16 hv = (_Float16)((rr[i] - m) * inv * grs[i] + brs[i]);
      dst[i] = hv;
      rnT[i * 280 + tid] = hv;
    }
  }
  __syncthreads();

  // GEMM1: scores_r = rn(256x128) @ wqB(128x16); 16 row-tiles, 2 per wave
  for (int t = 0; t < 2; ++t) {
    int mt = wave * 2 + t;
    v8f c = wmma_gemm_tile(rn, 136, wqB, 16, mt * 16, 0, 128, lane);
    int half = lane >> 4, lh = lane & 15;
    if (lh < 8) {
#pragma unroll
      for (int i = 0; i < 8; ++i)
        scF[(mt * 16 + half * 8 + i) * 8 + lh] += c[i];
    }
  }
  __syncthreads();

  // softmax: wave w handles head h=w over 256 keys
  {
    int h = wave;
    float vals[8];
    float mx = -3.4e38f;
#pragma unroll
    for (int j = 0; j < 8; ++j) {
      float sv = scF[(lane + 32 * j) * 8 + h] * SCALE_;
      vals[j] = sv;
      mx = fmaxf(mx, sv);
    }
    for (int off = 16; off; off >>= 1) mx = fmaxf(mx, __shfl_xor(mx, off, 32));
    float sum = 0.f;
#pragma unroll
    for (int j = 0; j < 8; ++j) { vals[j] = __expf(vals[j] - mx); sum += vals[j]; }
    for (int off = 16; off; off >>= 1) sum += __shfl_xor(sum, off, 32);
    float rinv = 1.f / sum;
#pragma unroll
    for (int j = 0; j < 8; ++j) {
      float p = vals[j] * rinv;
      int kk = lane + 32 * j;              // kb = j, in-block = lane
      scF[kk * 8 + h] = p;
      scB[(size_t)(j * 16 + h) * BPAD_ + lane] = (_Float16)p;
      scB[(size_t)(j * 16 + 8 + h) * BPAD_ + lane] = (_Float16)0.f;
    }
  }
  __syncthreads();

  // GEMM2: sr = rn^T(128x256) @ scB(256x16); 8 c-tiles, one per wave
  {
    int c0 = wave * 16;
    v8f c = wmma_gemm_tile(rnT, 280, scB, 16, c0, 0, 256, lane);
    int half = lane >> 4, lh = lane & 15;
    if (lh < 8) {
#pragma unroll
      for (int i = 0; i < 8; ++i) srF[(c0 + half * 8 + i) * 8 + lh] = c[i];
    }
  }
  __syncthreads();

  // epilogue: out[h,d] = bvr + sum_k score*v + sum_c sr*Wvr
  if (tid < 128) {
    int h = tid >> 4;
    const float* vb = v_ws + (size_t)bIdx * LK_ * D_ + tid;
    float acc = bvr[tid];
    for (int kk = 0; kk < LK_; ++kk) acc += scF[kk * 8 + h] * vb[(size_t)kk * D_];
    const float* wv = Wvr + tid;
    for (int c2 = 0; c2 < HID_; ++c2) acc += srF[c2 * 8 + h] * wv[(size_t)c2 * D_];
    attn_ws[((size_t)bIdx * LQ_ + iq) * D_ + tid] = acc;
  }
}

// ---- Kernel 4: gate + Wo + post-LN + FFN, 64 rows/WG -----------------------
// smem (183296 B): cat 64x264 h @0 (33792, reused as ffb f32 at end)
//  wb packed 8*128*40 h @33792 (81920) ; mix 64x136 h @115712 (17408)
//  chA 64x136 h @133120 (17408) ; hb 64x128 f @150528 (32768)
__global__ void k_post(const float* __restrict__ attn_ws, const float* __restrict__ xd_ws,
                       const float* __restrict__ s_ws,
                       const float* __restrict__ Wg, const float* __restrict__ bg,
                       const float* __restrict__ Wo, const float* __restrict__ bo,
                       const float* __restrict__ W1, const float* __restrict__ b1,
                       const float* __restrict__ W2, const float* __restrict__ b2,
                       const float* __restrict__ gpost, const float* __restrict__ bpost,
                       const float* __restrict__ gffpre, const float* __restrict__ bffpre,
                       const float* __restrict__ gffpost, const float* __restrict__ bffpost,
                       float* __restrict__ outp) {
  extern __shared__ char smem[];
  _Float16* cat = (_Float16*)smem;
  _Float16* wb  = (_Float16*)(smem + 33792);
  _Float16* mix = (_Float16*)(smem + 115712);
  _Float16* chA = (_Float16*)(smem + 133120);
  float* hb  = (float*)(smem + 150528);
  float* ffb = (float*)smem;   // reuse cat region after GEMM1

  int row0 = blockIdx.x * 64;
  int tid = threadIdx.x, lane = tid & 31, wave = tid >> 5;
  int half = lane >> 4, lh = lane & 15;

  for (int i = tid; i < 64 * 128; i += 256) {
    int rr = i >> 7, cc = i & 127;
    cat[rr * 264 + cc]       = (_Float16)attn_ws[(size_t)(row0 + rr) * D_ + cc];
    cat[rr * 264 + 128 + cc] = (_Float16)xd_ws[(size_t)(row0 + rr) * D_ + cc];
  }
  stage_w(wb, Wg, 256 * 128, tid);
  __syncthreads();

  // GEMM1: gate = sigmoid(cat @ Wg + bg); mix = out + g*(s-out)
  for (int t = 0; t < 4; ++t) {
    int id = wave * 4 + t, mt = id >> 3, nt = id & 7, col = nt * 16 + lh;
    v8f c = wmma_gemm_tile(cat, 264, wb, 128, mt * 16, nt * 16, 256, lane);
#pragma unroll
    for (int i = 0; i < 8; ++i) {
      int rl = mt * 16 + half * 8 + i, rg = row0 + rl;
      float gv = 1.f / (1.f + __expf(-(c[i] + bg[col])));
      float ov = attn_ws[(size_t)rg * D_ + col];
      float sv = s_ws[(size_t)rg * D_ + col];
      mix[rl * 136 + col] = (_Float16)(ov + gv * (sv - ov));
    }
  }
  __syncthreads();
  stage_w(wb, Wo, 128 * 128, tid);
  __syncthreads();
  // GEMM2: attn = mix @ Wo + bo
  for (int t = 0; t < 4; ++t) {
    int id = wave * 4 + t, mt = id >> 3, nt = id & 7, col = nt * 16 + lh;
    v8f c = wmma_gemm_tile(mix, 136, wb, 128, mt * 16, nt * 16, 128, lane);
#pragma unroll
    for (int i = 0; i < 8; ++i) {
      int rl = mt * 16 + half * 8 + i;
      hb[rl * 128 + col] = c[i] + bo[col];
    }
  }
  __syncthreads();
  // post-LN + residual; ffpre-LN -> mix
  if (tid < 64) {
    float* ar = hb + tid * 128;
    float m = 0.f;
    for (int i = 0; i < 128; ++i) m += ar[i];
    m *= (1.f / 128.f);
    float v2 = 0.f;
    for (int i = 0; i < 128; ++i) { float d = ar[i] - m; v2 += d * d; }
    float inv = rsqrtf(v2 * (1.f / 128.f) + 1e-5f);
    const float* xr = xd_ws + (size_t)(row0 + tid) * D_;
    float hm = 0.f;
    for (int i = 0; i < 128; ++i) {
      float hv = xr[i] + (ar[i] - m) * inv * gpost[i] + bpost[i];
      ar[i] = hv;
      hm += hv;
    }
    hm *= (1.f / 128.f);
    float hv2 = 0.f;
    for (int i = 0; i < 128; ++i) { float d = ar[i] - hm; hv2 += d * d; }
    float hinv = rsqrtf(hv2 * (1.f / 128.f) + 1e-5f);
    _Float16* mr = mix + tid * 136;
    for (int i = 0; i < 128; ++i)
      mr[i] = (_Float16)((ar[i] - hm) * hinv * gffpre[i] + bffpre[i]);
  }

  // FFN: relu(ffx@W1+b1)@W2 accumulated in registers across 4 K-chunks
  v8f acc[4];
#pragma unroll
  for (int t = 0; t < 4; ++t)
#pragma unroll
    for (int j = 0; j < 8; ++j) acc[t][j] = 0.f;

  for (int cc = 0; cc < 4; ++cc) {
    __syncthreads();
    for (int i = tid; i < 128 * 128; i += 256) {
      int kk = i >> 7, nn = i & 127;
      wb[(size_t)((kk >> 5) * 128 + nn) * BPAD_ + (kk & 31)] =
          (_Float16)W1[(size_t)kk * 512 + cc * 128 + nn];
    }
    __syncthreads();
    for (int t = 0; t < 4; ++t) {
      int id = wave * 4 + t, mt = id >> 3, nt = id & 7, col = nt * 16 + lh;
      v8f c = wmma_gemm_tile(mix, 136, wb, 128, mt * 16, nt * 16, 128, lane);
#pragma unroll
      for (int i = 0; i < 8; ++i) {
        int rl = mt * 16 + half * 8 + i;
        float rv = c[i] + b1[cc * 128 + col];
        chA[rl * 136 + col] = (_Float16)fmaxf(rv, 0.f);
      }
    }
    __syncthreads();
    for (int i = tid; i < 128 * 128; i += 256) {
      int kk = i >> 7, nn = i & 127;
      wb[(size_t)((kk >> 5) * 128 + nn) * BPAD_ + (kk & 31)] =
          (_Float16)W2[(size_t)(cc * 128 + kk) * 128 + nn];
    }
    __syncthreads();
    for (int t = 0; t < 4; ++t) {
      int id = wave * 4 + t, mt = id >> 3, nt = id & 7;
#pragma unroll
      for (int k0 = 0; k0 < 128; k0 += 32) {
        v16h a = load_a_frag(chA, mt * 16, k0, 136, lane);
        v16h b = load_b_packed(wb, k0 >> 5, nt * 16, 128, lane);
        acc[t] = __builtin_amdgcn_wmma_f32_16x16x32_f16(false, a, false, b,
                                                        (short)0, acc[t], false, false);
      }
    }
  }
  __syncthreads();
  for (int t = 0; t < 4; ++t) {
    int id = wave * 4 + t, mt = id >> 3, nt = id & 7, col = nt * 16 + lh;
#pragma unroll
    for (int i = 0; i < 8; ++i) {
      int rl = mt * 16 + half * 8 + i;
      ffb[rl * 128 + col] = acc[t][i] + b2[col];
    }
  }
  __syncthreads();
  // final: out = h + LN(ff)
  if (tid < 64) {
    const float* fr = ffb + tid * 128;
    float m = 0.f;
    for (int i = 0; i < 128; ++i) m += fr[i];
    m *= (1.f / 128.f);
    float v2 = 0.f;
    for (int i = 0; i < 128; ++i) { float d = fr[i] - m; v2 += d * d; }
    float inv = rsqrtf(v2 * (1.f / 128.f) + 1e-5f);
    float* orow = outp + (size_t)(row0 + tid) * D_;
    const float* hr = hb + tid * 128;
    for (int i = 0; i < 128; ++i)
      orow[i] = hr[i] + (fr[i] - m) * inv * gffpost[i] + bffpost[i];
  }
}

// ---- host launcher ---------------------------------------------------------
extern "C" void kernel_launch(void* const* d_in, const int* in_sizes, int n_in,
                              void* d_out, int out_size, void* d_ws, size_t ws_size,
                              hipStream_t stream) {
  (void)in_sizes; (void)n_in; (void)out_size; (void)ws_size;
  const float* x_src = (const float*)d_in[0];
  const float* x_dst = (const float*)d_in[1];
  const float* rten  = (const float*)d_in[2];
  // d_in[3..5]: src/dst/edge masks -- reference discards their effect
  const float* Wq  = (const float*)d_in[6];
  const float* bq  = (const float*)d_in[7];
  const float* Wk  = (const float*)d_in[8];
  const float* Wv  = (const float*)d_in[9];
  const float* bv  = (const float*)d_in[10];
  const float* Wkr = (const float*)d_in[11];
  const float* Wvr = (const float*)d_in[12];
  const float* bvr = (const float*)d_in[13];
  const float* Ws  = (const float*)d_in[14];
  const float* bs  = (const float*)d_in[15];
  const float* Wg  = (const float*)d_in[16];
  const float* bg  = (const float*)d_in[17];
  const float* Wo  = (const float*)d_in[18];
  const float* bo  = (const float*)d_in[19];
  const float* W1  = (const float*)d_in[20];
  const float* b1  = (const float*)d_in[21];
  const float* W2  = (const float*)d_in[22];
  const float* b2  = (const float*)d_in[23];
  const float* ln_src_g = (const float*)d_in[24];
  const float* ln_src_b = (const float*)d_in[25];
  const float* ln_dst_g = (const float*)d_in[26];
  const float* ln_dst_b = (const float*)d_in[27];
  const float* ln_r_g   = (const float*)d_in[28];
  const float* ln_r_b   = (const float*)d_in[29];
  const float* ln_post_g   = (const float*)d_in[30];
  const float* ln_post_b   = (const float*)d_in[31];
  const float* ln_ffpre_g  = (const float*)d_in[32];
  const float* ln_ffpre_b  = (const float*)d_in[33];
  const float* ln_ffpost_g = (const float*)d_in[34];
  const float* ln_ffpost_b = (const float*)d_in[35];

  float* ws = (float*)d_ws;
  float* q_ws    = ws;
  float* k_ws    = ws + 262144;
  float* v_ws    = ws + 524288;
  float* s_ws    = ws + 786432;
  float* xd_ws   = ws + 1048576;
  float* attn_ws = ws + 1310720;
  float* outp = (float*)d_out;

  k_proj_src<<<32, 256, 58368, stream>>>(x_src, Wk, Wv, bv, ln_src_g, ln_src_b,
                                         k_ws, v_ws);
  k_proj_dst<<<32, 256, 58368, stream>>>(x_dst, Wq, bq, Ws, bs, ln_dst_g, ln_dst_b,
                                         q_ws, s_ws, xd_ws);
  k_attn<<<B_ * LQ_, 256, 170496, stream>>>(rten, q_ws, k_ws, v_ws, Wkr, Wvr, bvr,
                                            ln_r_g, ln_r_b, attn_ws);
  k_post<<<32, 256, 183296, stream>>>(attn_ws, xd_ws, s_ws, Wg, bg, Wo, bo,
                                      W1, b1, W2, b2, ln_post_g, ln_post_b,
                                      ln_ffpre_g, ln_ffpre_b, ln_ffpost_g,
                                      ln_ffpost_b, outp);
}
